// MultiRelGraphLayer_25520695673302
// MI455X (gfx1250) — matrix operand order repaired and checked by reading, not compiled, not run on hardware
//
#include <hip/hip_runtime.h>
#include <stdint.h>

typedef __bf16 bf16_t;
typedef __attribute__((ext_vector_type(16))) __bf16 v16bf;
typedef __attribute__((ext_vector_type(8)))  __bf16 v8bf;
typedef __attribute__((ext_vector_type(8)))  float  v8f;
typedef unsigned int u32;
typedef __attribute__((ext_vector_type(4))) u32 v4u;
typedef __attribute__((ext_vector_type(4))) int v4i;
typedef __attribute__((ext_vector_type(8))) int v8i;

#define N_NODES 10000
#define E_EDGES 640000
#define D_IN    128
#define D_HID   128
#define H_HEADS 8
#define K_TOT   384          // 3 * D_IN
#define LEAKY   0.01f
#define RRELU   0.2291666666666667f

// Padded LDS layout for W tile written by TDM:
//  row = 384 bf16 = 768B; TDM pad: +16B after every 256B -> row stride 816B = 408 elems
#define W_ROW_ELEMS 408
#define W_LDS_BYTES (128 * W_ROW_ELEMS * 2)   // 104448

// ---------------- float atomic max via int/uint monotonic trick ----------------
__device__ __forceinline__ void atomicMaxF(float* addr, float v) {
  if (v >= 0.0f) atomicMax((int*)addr, __float_as_int(v));
  else           atomicMin((unsigned int*)addr, __float_as_uint(v));
}

// -------- Tensor Data Mover: DMA bf16 W tile (128 rows x 384 elems) into LDS --------
// D# per CDNA5 ISA. 2D tensor = tile, data_size=2B,
// pad_enable: interval 64 dwords (256B), amount 4 dwords (16B) -> row stride 816B.
__device__ __forceinline__ void tdm_load_w(const bf16_t* wt_global, bf16_t* sW) {
  const unsigned long long ga = (unsigned long long)(uintptr_t)wt_global;
  const u32 lds = (u32)(uintptr_t)sW;   // low 32 bits = LDS byte offset
  v4u g0 = { 1u,                        // count=1 (valid user descriptor)
             lds,                       // lds_addr
             (u32)ga,                   // global_addr[31:0]
             (u32)((ga >> 32) & 0x01FFFFFFull) | (2u << 30) };  // ga[56:32] | type=2
  v8i g1 = { (int)((1u << 16)          // data_size = 2 bytes
                 | (1u << 20)          // pad_enable
                 | (5u << 22)          // pad_interval: 64 dwords = 256B
                 | (3u << 25)),        // pad_amount: 4 dwords = 16B
             (int)(384u << 16),        // tensor_dim0 = 384 (bits 79:48, low part)
             (int)(128u << 16),        // tensor_dim1 = 128 (bits 111:80, low part)
             (int)(384u << 16),        // tile_dim0 = 384 (bits 127:112)
             128,                      // tile_dim1 = 128 (bits 143:128)
             384,                      // tensor_dim0_stride = 384 (bits 207:160, low)
             0, 0 };
  v4i g2 = { 0, 0, 0, 0 };
  v4i g3 = { 0, 0, 0, 0 };
#if defined(__clang_major__) && (__clang_major__ >= 23)
  v8i g4 = { 0, 0, 0, 0, 0, 0, 0, 0 };
  __builtin_amdgcn_tensor_load_to_lds(g0, g1, g2, g3, g4, 0);
#else
  __builtin_amdgcn_tensor_load_to_lds(g0, g1, g2, g3, 0);
#endif
  __builtin_amdgcn_s_wait_tensorcnt(0);
}

// ---------------- A-fragment loader (16-bit A 16x32, contiguous k per lane half) ----------------
// lanes 0-15: elems 0..15 -> K=kb+0..15
// lanes16-31: elems 0..15 -> K=kb+16..31
__device__ __forceinline__ v16bf load_a_frag(const float* __restrict__ rp, int b1, int b2) {
  const float4 a0 = *(const float4*)(rp + b1);
  const float4 a1 = *(const float4*)(rp + b1 + 4);
  const float4 a2 = *(const float4*)(rp + b2);
  const float4 a3 = *(const float4*)(rp + b2 + 4);
  v16bf af;
  af[0]=(bf16_t)a0.x; af[1]=(bf16_t)a0.y; af[2] =(bf16_t)a0.z; af[3] =(bf16_t)a0.w;
  af[4]=(bf16_t)a1.x; af[5]=(bf16_t)a1.y; af[6] =(bf16_t)a1.z; af[7] =(bf16_t)a1.w;
  af[8]=(bf16_t)a2.x; af[9]=(bf16_t)a2.y; af[10]=(bf16_t)a2.z; af[11]=(bf16_t)a2.w;
  af[12]=(bf16_t)a3.x; af[13]=(bf16_t)a3.y; af[14]=(bf16_t)a3.z; af[15]=(bf16_t)a3.w;
  return af;
}

// ------------- per-wave 16x128 edge GEMM: trip_hid tile in C regs -------------
// sW: bf16 W_trip transposed in LDS, padded rows of W_ROW_ELEMS
__device__ __forceinline__ void edge_gemm(const float* __restrict__ p0,
                                          const float* __restrict__ p1,
                                          const float* __restrict__ p2,
                                          const bf16_t* sW,
                                          int lane16, int half, v8f c[8]) {
  #pragma unroll
  for (int t = 0; t < 8; ++t) { v8f z = {}; c[t] = z; }
  #pragma unroll 1
  for (int kk = 0; kk < 12; ++kk) {
    const int seg = kk >> 2;
    const float* rp = (seg == 0) ? p0 : ((seg == 1) ? p1 : p2);
    const int kb = (kk & 3) * 32;
    const int b1 = kb + (half ? 16 : 0);
    const int b2 = kb + 8 + (half ? 16 : 0);
    const v16bf af = load_a_frag(rp, b1, b2);
    const int kcol = kk * 32 + (half ? 16 : 0);       // B: lanes<16 K 0..15, >=16 K 16..31
    const int kpad = kcol + ((kcol >> 7) << 3);       // +8 elems per 128-elem pad interval
    #pragma unroll
    for (int t = 0; t < 8; ++t) {
      const int base = (t * 16 + lane16) * W_ROW_ELEMS + kpad;
      const v8bf lo = *(const v8bf*)(sW + base);      // ds_load_b128
      const v8bf hi = *(const v8bf*)(sW + base + 8);
      v16bf bfr;
      #pragma unroll
      for (int j = 0; j < 8; ++j) { bfr[j] = lo[j]; bfr[j + 8] = hi[j]; }
      c[t] = __builtin_amdgcn_wmma_f32_16x16x32_bf16(
          false, af, false, bfr, (short)0, c[t], false, false);
    }
  }
}

// ---------------- kernel 1: W_trip -> bf16 transposed in workspace ----------------
__global__ void prep_wt_kernel(const float* __restrict__ W_trip, bf16_t* __restrict__ wt) {
  int i = blockIdx.x * 256 + threadIdx.x;
  if (i < D_HID * K_TOT) {
    int n = i / K_TOT, k = i % K_TOT;
    wt[i] = (bf16_t)W_trip[k * D_HID + n];
  }
}

// ---------------- kernel 2: init segmax/segsum/accum ----------------
__global__ void init_kernel(float* __restrict__ segmax, float* __restrict__ segsum,
                            float* __restrict__ accum) {
  int i = blockIdx.x * 256 + threadIdx.x;
  if (i < N_NODES * H_HEADS) { segmax[i] = -__builtin_inff(); segsum[i] = 0.0f; }
  if (i < N_NODES * D_HID) accum[i] = 0.0f;
}

// ---------------- kernel 3: edge GEMM + scores + segment max ----------------
__global__ __launch_bounds__(128) void edge_score_kernel(
    const float* __restrict__ node_feats, const float* __restrict__ edge_feats,
    const int* __restrict__ src, const int* __restrict__ dst,
    const bf16_t* __restrict__ wt, const float* __restrict__ b_trip,
    const float* __restrict__ W_score, const float* __restrict__ b_score,
    float* __restrict__ scores, float* __restrict__ segmax) {
  extern __shared__ char smem[];
  bf16_t* sW  = (bf16_t*)smem;                                  // W_LDS_BYTES
  float*  sH  = (float*)(smem + W_LDS_BYTES);                   // 4*16*132*4 = 33792
  float*  sWs = (float*)(smem + W_LDS_BYTES + 33792);           // 4096
  float*  sBt = (float*)(smem + W_LDS_BYTES + 33792 + 4096);    // 512
  float*  sBs = (float*)(smem + W_LDS_BYTES + 33792 + 4096 + 512);

  const int tid = threadIdx.x;
  if (tid < 32) tdm_load_w(wt, sW);     // wave 0: TDM DMA + s_wait_tensorcnt
  for (int i = tid; i < D_HID * H_HEADS; i += 128) sWs[i] = W_score[i];
  if (tid < D_HID) sBt[tid] = b_trip[tid];
  if (tid < H_HEADS) sBs[tid] = b_score[tid];
  __syncthreads();

  const int wave = tid >> 5, lane = tid & 31;
  const int lane16 = lane & 15, half = lane >> 4;
  const int e0 = (blockIdx.x * 4 + wave) * 16;  // grid sized so e0+15 < E
  const int e  = e0 + lane16;

  const float* p0 = edge_feats + (size_t)e * D_IN;
  const float* p1 = node_feats + (size_t)src[e] * D_IN;
  const float* p2 = node_feats + (size_t)dst[e] * D_IN;
  __builtin_prefetch(p0, 0, 1);
  __builtin_prefetch(p1, 0, 1);

  v8f c[8];
  edge_gemm(p0, p1, p2, sW, lane16, half, c);

  // stage trip_hid (+bias): C layout: lane16 = col, VGPR v -> row v + 8*half
  #pragma unroll
  for (int t = 0; t < 8; ++t) {
    const int n = t * 16 + lane16;
    #pragma unroll
    for (int v = 0; v < 8; ++v)
      sH[(wave * 16 + v + half * 8) * 132 + n] = c[t][v] + sBt[n];
  }
  __syncthreads();

  // scores: 16 edges x 8 heads per wave, 4 items per lane
  #pragma unroll
  for (int q = 0; q < 4; ++q) {
    const int item = q * 32 + lane;
    const int m = item >> 3, h = item & 7;
    float s = sBs[h];
    #pragma unroll 4
    for (int k = 0; k < D_HID; ++k) s += sH[(wave * 16 + m) * 132 + k] * sWs[k * H_HEADS + h];
    s = (s >= 0.0f) ? s : LEAKY * s;
    const int eg = e0 + m;
    scores[(size_t)eg * H_HEADS + h] = s;
    atomicMaxF(&segmax[(size_t)dst[eg] * H_HEADS + h], s);
  }
}

// ---------------- kernel 4: ex = exp(score - max), segment sum ----------------
__global__ void softmax_norm_kernel(const int* __restrict__ dst,
                                    float* __restrict__ scores,
                                    const float* __restrict__ segmax,
                                    float* __restrict__ segsum) {
  size_t i = (size_t)blockIdx.x * 256 + threadIdx.x;
  if (i >= (size_t)E_EDGES * H_HEADS) return;
  const int e = (int)(i >> 3), h = (int)(i & 7);
  const int d = dst[e];
  const float ex = __expf(scores[i] - segmax[(size_t)d * H_HEADS + h]);
  scores[i] = ex;
  atomicAdd(&segsum[(size_t)d * H_HEADS + h], ex);
}

// ---------------- kernel 5: recompute GEMM, weighted scatter-sum ----------------
__global__ __launch_bounds__(128) void aggregate_kernel(
    const float* __restrict__ node_feats, const float* __restrict__ edge_feats,
    const int* __restrict__ src, const int* __restrict__ dst,
    const bf16_t* __restrict__ wt, const float* __restrict__ b_trip,
    const float* __restrict__ ex, const float* __restrict__ segsum,
    float* __restrict__ accum) {
  extern __shared__ char smem[];
  bf16_t* sW   = (bf16_t*)smem;                       // W_LDS_BYTES
  float*  sWgt = (float*)(smem + W_LDS_BYTES);        // 4*16*4 = 256
  int*    sDst = (int*)(smem + W_LDS_BYTES + 256);    // 256

  const int tid = threadIdx.x;
  if (tid < 32) tdm_load_w(wt, sW);
  __syncthreads();

  const int wave = tid >> 5, lane = tid & 31;
  const int lane16 = lane & 15, half = lane >> 4;
  const int e0 = (blockIdx.x * 4 + wave) * 16;
  const int e  = e0 + lane16;

  const float* p0 = edge_feats + (size_t)e * D_IN;
  const float* p1 = node_feats + (size_t)src[e] * D_IN;
  const float* p2 = node_feats + (size_t)dst[e] * D_IN;
  __builtin_prefetch(p0, 0, 1);

  v8f c[8];
  edge_gemm(p0, p1, p2, sW, lane16, half, c);

  if (half == 0) {   // lanes 0..15 compute per-edge mean attention weight
    const int d = dst[e];
    float wsum = 0.0f;
    #pragma unroll
    for (int h = 0; h < H_HEADS; ++h)
      wsum += ex[(size_t)e * H_HEADS + h] / segsum[(size_t)d * H_HEADS + h];
    sWgt[wave * 16 + lane16] = wsum * (1.0f / H_HEADS);
    sDst[wave * 16 + lane16] = d;
  }
  __syncthreads();

  #pragma unroll
  for (int t = 0; t < 8; ++t) {
    const int n = t * 16 + lane16;
    const float bt = b_trip[n];
    #pragma unroll
    for (int v = 0; v < 8; ++v) {
      const int m = v + half * 8;
      const float val = sWgt[wave * 16 + m] * (c[t][v] + bt);
      atomicAdd(&accum[(size_t)sDst[wave * 16 + m] * D_HID + n], val);
    }
  }
}

// ---------------- kernel 6: self GEMM + add + RReLU ----------------
__global__ __launch_bounds__(128) void self_out_kernel(
    const float* __restrict__ node_feats, const float* __restrict__ W_self,
    const float* __restrict__ b_self, const float* __restrict__ accum,
    float* __restrict__ out) {
  __shared__ float sRow[D_IN];
  const int i = blockIdx.x, j = threadIdx.x;
  sRow[j] = node_feats[(size_t)i * D_IN + j];
  __syncthreads();
  float acc = b_self[j];
  #pragma unroll 4
  for (int k = 0; k < D_IN; ++k) acc += sRow[k] * W_self[k * D_HID + j];
  const float x = acc + accum[(size_t)i * D_HID + j];
  out[(size_t)i * D_HID + j] = (x >= 0.0f) ? x : RRELU * x;
}

// ---------------- launch ----------------
extern "C" void kernel_launch(void* const* d_in, const int* in_sizes, int n_in,
                              void* d_out, int out_size, void* d_ws, size_t ws_size,
                              hipStream_t stream) {
  const float* node_feats = (const float*)d_in[0];
  const float* edge_feats = (const float*)d_in[1];
  const int*   src        = (const int*)d_in[2];
  const int*   dst        = (const int*)d_in[3];
  const float* W_trip     = (const float*)d_in[4];
  const float* b_trip     = (const float*)d_in[5];
  const float* W_score    = (const float*)d_in[6];
  const float* b_score    = (const float*)d_in[7];
  const float* W_self     = (const float*)d_in[8];
  const float* b_self     = (const float*)d_in[9];
  float* out = (float*)d_out;

  // workspace layout (floats)
  float* ws      = (float*)d_ws;
  float* scores  = ws;                                   // E*8
  float* segmax  = scores + (size_t)E_EDGES * H_HEADS;   // N*8
  float* segsum  = segmax + (size_t)N_NODES * H_HEADS;   // N*8
  float* accum   = segsum + (size_t)N_NODES * H_HEADS;   // N*128
  bf16_t* wt     = (bf16_t*)(accum + (size_t)N_NODES * D_HID);  // 128*384 bf16

  (void)in_sizes; (void)n_in; (void)out_size; (void)ws_size;

  prep_wt_kernel<<<(D_HID * K_TOT + 255) / 256, 256, 0, stream>>>(W_trip, wt);
  init_kernel<<<(N_NODES * D_HID + 255) / 256, 256, 0, stream>>>(segmax, segsum, accum);

  const int edge_blocks = E_EDGES / 64;   // 4 waves x 16 edges per block
  const size_t edge_smem = W_LDS_BYTES + 33792 + 4096 + 512 + 64;
  edge_score_kernel<<<edge_blocks, 128, edge_smem, stream>>>(
      node_feats, edge_feats, src, dst, wt, b_trip, W_score, b_score, scores, segmax);

  softmax_norm_kernel<<<(E_EDGES * H_HEADS + 255) / 256, 256, 0, stream>>>(
      dst, scores, segmax, segsum);

  const size_t agg_smem = W_LDS_BYTES + 256 + 256;
  aggregate_kernel<<<edge_blocks, 128, agg_smem, stream>>>(
      node_feats, edge_feats, src, dst, wt, b_trip, scores, segsum, accum);

  self_out_kernel<<<N_NODES, 128, 0, stream>>>(node_feats, W_self, b_self, accum, out);
}
